// HybridMoE_120259085108
// MI455X (gfx1250) — matrix operand order repaired and checked by reading, not compiled, not run on hardware
//
#include <hip/hip_runtime.h>
#include <hip/hip_bf16.h>

#define TOKENS 8192
#define HIDDEN 1024
#define INTER  2816
#define NEXP   16
#define TOPK   2
#define MT     32          // tokens per block (M tile)

// padded LDS strides (elements); bases stay 16B-aligned, banks spread.
#define XP 1032            // X tile row stride   (2064 B/row)
#define HP 2824            // h tile row stride   (5648 B/row)
#define WP 72              // weight col stride for K=64 tiles (144 B/col)

typedef __attribute__((ext_vector_type(16))) __bf16 v16bf;
typedef __attribute__((ext_vector_type(8)))  __bf16 v8bf;
typedef __attribute__((ext_vector_type(4)))  __bf16 v4bf;
typedef __attribute__((ext_vector_type(2)))  __bf16 v2bf;
typedef __attribute__((ext_vector_type(8)))  float  v8f;

union Frag { v16bf v; v8bf h[2]; };

// A fragment (16x32 bf16): lane = half*16 + m; row m, K contiguous:
// half 0 -> {k..k+15}; half 1 -> {k+16..k+31}
__device__ __forceinline__ v16bf load_a_frag(const __bf16* rowbase, int k, int half) {
  Frag f;
  f.h[0] = *(const v8bf*)(rowbase + k + half * 16);
  f.h[1] = *(const v8bf*)(rowbase + k + half * 16 + 8);
  return f.v;
}

// B fragment (32x16 bf16) from a column-major LDS tile: lane = half*16 + n;
// column n, K = half*16 .. half*16+15 contiguous down the column.
__device__ __forceinline__ v16bf load_b_frag(const __bf16* colbase, int half) {
  Frag f;
  f.h[0] = *(const v8bf*)(colbase + half * 16);
  f.h[1] = *(const v8bf*)(colbase + half * 16 + 8);
  return f.v;
}

// ---- split staging of a 64(K) x 64(N) fp32 tile (software pipeline) -------
// wload: issue global b128 loads into registers, NO wait; plus a prefetch
// two tiles ahead so L2 stays warm at streaming rate.
__device__ __forceinline__ void wload(const float* __restrict__ W, int ldw,
                                      int k0, int n0, int kmax, int tid,
                                      float4 r[4]) {
#pragma unroll
  for (int i = 0; i < 2; ++i) {
    int w  = tid + (i << 8);       // 0..511
    int kp = w >> 4;               // k-pair 0..31 (k = 2*kp)
    int c4 = w & 15;               // float4 column -> n = 4*c4
    const float* p0 = W + (size_t)(k0 + 2 * kp) * ldw + n0 + 4 * c4;
    r[2 * i + 0] = *(const float4*)p0;
    r[2 * i + 1] = *(const float4*)(p0 + ldw);
    if (k0 + 128 < kmax) __builtin_prefetch(p0 + (size_t)128 * ldw, 0, 0);
  }
}

// wstore: convert (v_cvt_pk_bf16_f32) and write packed K-pairs as b32 LDS
// stores into the column-major tile. The s_wait_loadcnt lands here, a full
// K-tile of WMMA work after the loads were issued.
__device__ __forceinline__ void wstore(const float4 r[4], __bf16* dst, int tid) {
#pragma unroll
  for (int i = 0; i < 2; ++i) {
    int w  = tid + (i << 8);
    int kp = w >> 4;
    int c4 = w & 15;
    const float4 a = r[2 * i + 0];
    const float4 b = r[2 * i + 1];
    __bf16* d = dst + (4 * c4) * WP + 2 * kp;
    v2bf t0 = { (__bf16)a.x, (__bf16)b.x };
    v2bf t1 = { (__bf16)a.y, (__bf16)b.y };
    v2bf t2 = { (__bf16)a.z, (__bf16)b.z };
    v2bf t3 = { (__bf16)a.w, (__bf16)b.w };
    *(v2bf*)(d + 0 * WP) = t0;
    *(v2bf*)(d + 1 * WP) = t1;
    *(v2bf*)(d + 2 * WP) = t2;
    *(v2bf*)(d + 3 * WP) = t3;
  }
}

// ---------------------------------------------------------------------------
__global__ void zero_out_kernel(float4* __restrict__ p, int n4) {
  int i = blockIdx.x * blockDim.x + threadIdx.x;
  if (i < n4) p[i] = make_float4(0.f, 0.f, 0.f, 0.f);
}

// ---------------------------------------------------------------------------
// Router: one wave per token; 16 logits; top-2 softmax weights.
// ---------------------------------------------------------------------------
__global__ __launch_bounds__(256) void router_kernel(
    const float* __restrict__ X, const float* __restrict__ R,
    int* __restrict__ topk_i, float* __restrict__ topk_w) {
  const int lane = threadIdx.x & 31;
  const int wid  = threadIdx.x >> 5;
  const int t    = blockIdx.x * 8 + wid;

  float acc[NEXP];
#pragma unroll
  for (int e = 0; e < NEXP; ++e) acc[e] = 0.f;

  const float* x = X + (size_t)t * HIDDEN;
  for (int k = lane; k < HIDDEN; k += 32) {
    float xv = x[k];
    const float4* rp = (const float4*)(R + k * NEXP);
#pragma unroll
    for (int q = 0; q < 4; ++q) {
      float4 rv = rp[q];
      acc[4 * q + 0] += xv * rv.x;
      acc[4 * q + 1] += xv * rv.y;
      acc[4 * q + 2] += xv * rv.z;
      acc[4 * q + 3] += xv * rv.w;
    }
  }
#pragma unroll
  for (int e = 0; e < NEXP; ++e) {
#pragma unroll
    for (int off = 16; off > 0; off >>= 1) acc[e] += __shfl_xor(acc[e], off, 32);
  }
  if (lane == 0) {
    int i1 = 0; float l1 = acc[0];
    for (int e = 1; e < NEXP; ++e) if (acc[e] > l1) { l1 = acc[e]; i1 = e; }
    int i2 = (i1 == 0) ? 1 : 0; float l2 = acc[i2];
    for (int e = 0; e < NEXP; ++e)
      if (e != i1 && acc[e] > l2) { l2 = acc[e]; i2 = e; }
    float w1 = 1.f / (1.f + __expf(l2 - l1));   // == renormalized top-2 softmax
    topk_i[t * 2 + 0] = i1;
    topk_i[t * 2 + 1] = i2;
    topk_w[t * 2 + 0] = w1;
    topk_w[t * 2 + 1] = 1.f - w1;
  }
}

// ---------------------------------------------------------------------------
// Deterministic bucketing by expert (token order preserved, no atomics).
// ---------------------------------------------------------------------------
__global__ void gather_kernel(const int* __restrict__ topk_i,
                              const float* __restrict__ topk_w,
                              int* __restrict__ tok_list, float* __restrict__ tok_w,
                              int* __restrict__ e_off, int* __restrict__ t_off) {
  __shared__ int cnts[NEXP];
  __shared__ int offs[NEXP];
  const int lane = threadIdx.x;
  if (lane < NEXP) {
    int c = 0;
    for (int i = 0; i < TOKENS * TOPK; ++i) c += (topk_i[i] == lane) ? 1 : 0;
    cnts[lane] = c;
  }
  __syncthreads();
  if (lane == 0) {
    int run = 0, trun = 0;
    for (int e = 0; e < NEXP; ++e) {
      offs[e] = run;
      e_off[e] = run;
      t_off[e] = trun;
      run  += cnts[e];
      trun += (cnts[e] + MT - 1) / MT;
    }
    e_off[NEXP] = run;
    t_off[NEXP] = trun;
  }
  __syncthreads();
  if (lane < NEXP) {
    int pos = offs[lane];
    for (int i = 0; i < TOKENS * TOPK; ++i) {
      if (topk_i[i] == lane) {
        tok_list[pos] = i >> 1;
        tok_w[pos]    = topk_w[i];
        ++pos;
      }
    }
  }
}

// ---------------------------------------------------------------------------
// Fused expert FFN. Block = 32 tokens of one expert, 8 waves as 2(M) x 4(N).
// h[32 x INTER] lives entirely in LDS; weight tiles double-buffered and
// software-pipelined (loads for tile k+64 fly during tile k's WMMAs).
// Total LDS ~277 KB (CDNA5 WGP: 320 KB).
// ---------------------------------------------------------------------------
__global__ __launch_bounds__(256) void moe_ffn_kernel(
    const float* __restrict__ X,
    const float* __restrict__ Wg, const float* __restrict__ Wu,
    const float* __restrict__ Wd,
    const int* __restrict__ tok_list, const float* __restrict__ tok_w,
    const int* __restrict__ e_off, const int* __restrict__ t_off,
    float* __restrict__ out) {
  __shared__ __align__(16) __bf16 Xl[MT * XP];        //  66 048 B
  __shared__ __align__(16) __bf16 Hl[MT * HP];        // 180 736 B
  __shared__ __align__(16) __bf16 Wa[2][64 * WP];     // 2 x 9 216 B
  __shared__ __align__(16) __bf16 Wb[2][64 * WP];     // 2 x 9 216 B
  __shared__ int   tokl[MT];
  __shared__ float twl[MT];

  const int tid  = threadIdx.x;
  const int lane = tid & 31;
  const int wid  = tid >> 5;
  const int mw   = wid >> 2;        // 0..1 : 16-token half
  const int nw   = wid & 3;         // 0..3 : 16-col slice of the 64-chunk
  const int half = lane >> 4;
  const int nn   = lane & 15;

  const int b = blockIdx.x;
  if (b >= t_off[NEXP]) return;     // uniform per block
  int e = 0;
  while (e < NEXP - 1 && b >= t_off[e + 1]) ++e;
  const int mt    = b - t_off[e];
  const int abase = e_off[e] + mt * MT;
  const int aend  = e_off[e + 1];

  if (tid < MT) {
    int idx = abase + tid;
    bool valid = idx < aend;
    tokl[tid] = valid ? tok_list[idx] : -1;
    twl[tid]  = valid ? tok_w[idx] : 0.f;
  }
  __syncthreads();

  // ---- X tile (MT x HIDDEN) -> LDS bf16 (tail rows zero) -------------------
  for (int m = 0; m < MT; ++m) {
    int tok = tokl[m];
    float4 v = make_float4(0.f, 0.f, 0.f, 0.f);
    if (tok >= 0) v = *(const float4*)(X + (size_t)tok * HIDDEN + 4 * tid);
    v4bf bv = { (__bf16)v.x, (__bf16)v.y, (__bf16)v.z, (__bf16)v.w };
    *(v4bf*)(Xl + m * XP + 4 * tid) = bv;
  }

  const float* wg = Wg + (size_t)e * HIDDEN * INTER;
  const float* wu = Wu + (size_t)e * HIDDEN * INTER;
  const float* wd = Wd + (size_t)e * INTER * HIDDEN;

  const __bf16* arow = Xl + (size_t)(mw * 16 + nn) * XP;
  const __bf16* hrow = Hl + (size_t)(mw * 16 + nn) * HP;

  float4 rg[4], ru[4];

  // ---- gate/up: h[MT x INTER] in 64-wide N chunks, pipelined K=64 tiles ----
  for (int n0 = 0; n0 < INTER; n0 += 64) {
    v8f ag = {0.f, 0.f, 0.f, 0.f, 0.f, 0.f, 0.f, 0.f};
    v8f au = {0.f, 0.f, 0.f, 0.f, 0.f, 0.f, 0.f, 0.f};
    wload(wg, INTER, 0, n0, HIDDEN, tid, rg);
    wload(wu, INTER, 0, n0, HIDDEN, tid, ru);
    int cur = 0;
    for (int k0 = 0; k0 < HIDDEN; k0 += 64) {
      wstore(rg, Wa[cur], tid);          // waits the in-flight loads here
      wstore(ru, Wb[cur], tid);
      __syncthreads();
      if (k0 + 64 < HIDDEN) {            // issue next tile's loads, no wait
        wload(wg, INTER, k0 + 64, n0, HIDDEN, tid, rg);
        wload(wu, INTER, k0 + 64, n0, HIDDEN, tid, ru);
      }
      const __bf16* cg = Wa[cur] + (nw * 16 + nn) * WP;
      const __bf16* cu = Wb[cur] + (nw * 16 + nn) * WP;
#pragma unroll
      for (int ks = 0; ks < 64; ks += 32) {
        v16bf a  = load_a_frag(arow, k0 + ks, half);
        v16bf bg = load_b_frag(cg + ks, half);
        v16bf bu = load_b_frag(cu + ks, half);
        ag = __builtin_amdgcn_wmma_f32_16x16x32_bf16(false, a, false, bg, (short)0, ag, false, false);
        au = __builtin_amdgcn_wmma_f32_16x16x32_bf16(false, a, false, bu, (short)0, au, false, false);
      }
      cur ^= 1;
    }
    // silu(g)*u -> Hl.  C layout: VGPR r, lane -> M = half*8+r, N = nn.
    const int nbc = n0 + nw * 16 + nn;
#pragma unroll
    for (int r = 0; r < 8; ++r) {
      int m = mw * 16 + half * 8 + r;
      float g = ag[r], u = au[r];
      float s = __builtin_amdgcn_rcpf(1.f + __expf(-g));   // fast silu
      Hl[m * HP + nbc] = (__bf16)(g * s * u);
    }
  }
  __syncthreads();

  // ---- down: out_tile[MT x HIDDEN] = h @ Wd, weighted atomic scatter -------
  for (int n0 = 0; n0 < HIDDEN; n0 += 64) {
    v8f ac = {0.f, 0.f, 0.f, 0.f, 0.f, 0.f, 0.f, 0.f};
    wload(wd, HIDDEN, 0, n0, INTER, tid, rg);
    int cur = 0;
    for (int k0 = 0; k0 < INTER; k0 += 64) {
      wstore(rg, Wa[cur], tid);
      __syncthreads();
      if (k0 + 64 < INTER)
        wload(wd, HIDDEN, k0 + 64, n0, INTER, tid, rg);
      const __bf16* cd = Wa[cur] + (nw * 16 + nn) * WP;
#pragma unroll
      for (int ks = 0; ks < 64; ks += 32) {
        v16bf a  = load_a_frag(hrow, k0 + ks, half);
        v16bf bd = load_b_frag(cd + ks, half);
        ac = __builtin_amdgcn_wmma_f32_16x16x32_bf16(false, a, false, bd, (short)0, ac, false, false);
      }
      cur ^= 1;
    }
    const int col = n0 + nw * 16 + nn;
#pragma unroll
    for (int r = 0; r < 8; ++r) {
      int m = mw * 16 + half * 8 + r;
      int tok = tokl[m];
      if (tok >= 0)
        atomicAdd(out + (size_t)tok * HIDDEN + col, twl[m] * ac[r]);
    }
  }
}

// ---------------------------------------------------------------------------
extern "C" void kernel_launch(void* const* d_in, const int* in_sizes, int n_in,
                              void* d_out, int out_size, void* d_ws, size_t ws_size,
                              hipStream_t stream) {
  const float* X  = (const float*)d_in[0];   // [TOKENS, HIDDEN]
  const float* R  = (const float*)d_in[1];   // [HIDDEN, NEXP]
  const float* Wg = (const float*)d_in[2];   // [NEXP, HIDDEN, INTER]
  const float* Wu = (const float*)d_in[3];   // [NEXP, HIDDEN, INTER]
  const float* Wd = (const float*)d_in[4];   // [NEXP, INTER, HIDDEN]
  float* out = (float*)d_out;

  int*   topk_i  = (int*)d_ws;
  float* topk_w  = (float*)(topk_i + TOKENS * TOPK);
  int*   tok_lst = (int*)(topk_w + TOKENS * TOPK);
  float* tok_w   = (float*)(tok_lst + TOKENS * TOPK);
  int*   e_off   = (int*)(tok_w + TOKENS * TOPK);
  int*   t_off   = e_off + 32;

  const int n4 = TOKENS * HIDDEN / 4;
  zero_out_kernel<<<(n4 + 255) / 256, 256, 0, stream>>>((float4*)out, n4);
  router_kernel<<<TOKENS / 8, 256, 0, stream>>>(X, R, topk_i, topk_w);
  gather_kernel<<<1, 32, 0, stream>>>(topk_i, topk_w, tok_lst, tok_w, e_off, t_off);

  const int maxtiles = TOKENS * TOPK / MT + NEXP;  // 528: upper bound incl. tails
  moe_ffn_kernel<<<maxtiles, 256, 0, stream>>>(X, Wg, Wu, Wd, tok_lst, tok_w,
                                               e_off, t_off, out);
}
